// RoIRelationPredictor_5935644803770
// MI455X (gfx1250) — matrix operand order repaired and checked by reading, not compiled, not run on hardware
//
#include <hip/hip_runtime.h>

typedef __attribute__((ext_vector_type(16))) _Float16 v16h;
typedef __attribute__((ext_vector_type(8)))  _Float16 v8h;
typedef __attribute__((ext_vector_type(8)))  float    v8f;
typedef __attribute__((ext_vector_type(4)))  int      v4i;

#define N_OBJ    8192
#define HIDDEN   512
#define POOL     4096
#define N_REL    32768
#define N_RELCLS 51
#define NOBJCLS  151

// ---------- fragment helper: build v16h from two 16-byte segments ----------
__device__ __forceinline__ v16h frag16(const _Float16* p0, const _Float16* p1) {
    v8h lo = *(const v8h*)p0;
    v8h hi = *(const v8h*)p1;
    return __builtin_shufflevector(lo, hi, 0,1,2,3,4,5,6,7,8,9,10,11,12,13,14,15);
}

__device__ __forceinline__ v8f wmma_f16(v16h a, v16h b, v8f c) {
    return __builtin_amdgcn_wmma_f32_16x16x32_f16(false, a, false, b, (short)0, c, false, false);
}

// ---------- CDNA5 async global->LDS (ASYNCcnt path) ----------
__device__ __forceinline__ void async_b128(const _Float16* g, _Float16* l) {
#if __has_builtin(__builtin_amdgcn_global_load_async_to_lds_b128)
    __builtin_amdgcn_global_load_async_to_lds_b128(
        (__attribute__((address_space(1))) v4i*)(v4i*)g,
        (__attribute__((address_space(3))) v4i*)(v4i*)l, 0, 0);
#else
    unsigned lofs = (unsigned)(unsigned long long)l;
    asm volatile("global_load_async_to_lds_b128 %0, %1, off" :: "v"(lofs), "v"(g) : "memory");
#endif
}

__device__ __forceinline__ void wait_async0() {
#if __has_builtin(__builtin_amdgcn_s_wait_asynccnt)
    __builtin_amdgcn_s_wait_asynccnt(0);
    asm volatile("" ::: "memory");
#else
    asm volatile("s_wait_asynccnt 0x0" ::: "memory");
#endif
}

// ---------- prep kernels ----------
__global__ void f32_to_f16_kernel(const float* __restrict__ in, _Float16* __restrict__ out, int n) {
    int i = blockIdx.x * 256 + threadIdx.x;
    if (i < n) out[i] = (_Float16)in[i];
}

// in [R][C] fp32 -> out [C][R] f16
__global__ void transpose_f16_kernel(const float* __restrict__ in, _Float16* __restrict__ out, int R, int C) {
    int i = blockIdx.x * 256 + threadIdx.x;
    if (i < R * C) {
        int c = i / R, r = i - c * R;
        out[(size_t)c * R + r] = (_Float16)in[(size_t)r * C + c];
    }
}

// W_rel [4096][51] fp32 -> WrelT [64][4096] f16, rows 51..63 zero
__global__ void transpose_rel_kernel(const float* __restrict__ in, _Float16* __restrict__ out) {
    int i = blockIdx.x * 256 + threadIdx.x;
    if (i < 64 * POOL) {
        int c = i / POOL, k = i - c * POOL;
        out[i] = (c < N_RELCLS) ? (_Float16)in[(size_t)k * N_RELCLS + c] : (_Float16)0.f;
    }
}

// ---------- obj_preds = argmax(edge_ctx @ W_obj + b_obj) ----------
__global__ __launch_bounds__(256)
void obj_argmax_kernel(const float* __restrict__ edge_ctx, const float* __restrict__ W_obj,
                       const float* __restrict__ b_obj, int* __restrict__ preds) {
    __shared__ float rowv[HIDDEN];
    __shared__ float sval[256];
    __shared__ int   sidx[256];
    const int r = blockIdx.x, tid = threadIdx.x;
    rowv[tid]       = edge_ctx[(size_t)r * HIDDEN + tid];
    rowv[tid + 256] = edge_ctx[(size_t)r * HIDDEN + tid + 256];
    __syncthreads();
    float v = -INFINITY;
    if (tid < NOBJCLS) {
        float acc = b_obj[tid];
        for (int k = 0; k < HIDDEN; ++k) acc += rowv[k] * W_obj[(size_t)k * NOBJCLS + tid];
        v = acc;
    }
    sval[tid] = v; sidx[tid] = tid;
    __syncthreads();
    for (int s = 128; s > 0; s >>= 1) {
        if (tid < s) {
            float v2 = sval[tid + s]; int i2 = sidx[tid + s];
            if (v2 > sval[tid] || (v2 == sval[tid] && i2 < sidx[tid])) { sval[tid] = v2; sidx[tid] = i2; }
        }
        __syncthreads();
    }
    if (tid == 0) preds[r] = sidx[0];
}

// ---------- edge_rep = edge_ctx @ W_post_emb + b_post_emb  (f16 out [8192][1024]) ----------
#define HP (HIDDEN + 8)   // padded row to kill LDS bank conflicts

__global__ __launch_bounds__(128, 2)
void post_emb_kernel(const _Float16* __restrict__ ectx,   // [8192][512] f16
                     const _Float16* __restrict__ WembT,  // [1024][512] f16
                     const float* __restrict__ b_emb,     // [1024]
                     _Float16* __restrict__ erep)         // [8192][1024] f16
{
    __shared__ __align__(16) _Float16 Al[64][HP];
    const int tid = threadIdx.x;
    const int wave = tid >> 5, lane = tid & 31, l15 = lane & 15, hiq = lane >> 4;
    const int kb = hiq ? 8 : 0, kc = hiq ? 8 : 0;
    const int row0 = blockIdx.x * 64;
    const int n0   = blockIdx.y * 64;

    for (int idx = tid; idx < 64 * (HIDDEN / 8); idx += 128) {
        int r = idx >> 6, k = (idx & 63) * 8;
        async_b128(&ectx[(size_t)(row0 + r) * HIDDEN + k], &Al[r][k]);
    }
    wait_async0();
    __syncthreads();

    const int mband = wave * 16;
    const int arow  = mband + l15;
    v8f acc[4] = {};
    for (int ks = 0; ks < HIDDEN; ks += 32) {
        v16h af = frag16(&Al[arow][ks + kb], &Al[arow][ks + 16 + kb]);
        #pragma unroll
        for (int nt = 0; nt < 4; ++nt) {
            const _Float16* bp = &WembT[(size_t)(n0 + nt * 16 + l15) * HIDDEN + ks + kc];
            acc[nt] = wmma_f16(af, frag16(bp, bp + 16), acc[nt]);
        }
    }
    #pragma unroll
    for (int nt = 0; nt < 4; ++nt) {
        const int col = n0 + nt * 16 + l15;
        const float bias = b_emb[col];
        #pragma unroll
        for (int j = 0; j < 8; ++j) {
            const int m = j + 8 * hiq;
            erep[(size_t)(row0 + mband + m) * 1024 + col] = (_Float16)(acc[nt][j] + bias);
        }
    }
}

// ---------- fused:  gather -> @W_post_cat +b -> *union -> @W_rel +b_rel +freq_bias ----------
#define MT   64
#define KA   1024
#define KAP  (KA + 8)     // padded A row
#define PN   64
#define KSEG 256          // K per async-staged B segment
#define BNP  (KSEG + 8)   // padded B row
#define PNP  (PN + 8)     // padded P row

// stage one B segment (64 cols x 256 k, f16) asynchronously into LDS
__device__ __forceinline__ void stage_B(const _Float16* __restrict__ WcatT,
                                        _Float16 (*Bbuf)[BNP], int pc, int seg, int tid) {
    const _Float16* src = WcatT + (size_t)pc * PN * KA + seg * KSEG;
    for (int idx = tid; idx < PN * (KSEG / 8); idx += 128) {   // 2048 / 128 = 16 per thread
        int n = idx >> 5;
        int k = (idx & 31) * 8;
        async_b128(src + (size_t)n * KA + k, &Bbuf[n][k]);
    }
}

__global__ __launch_bounds__(128, 1)
void rel_fused_kernel(const _Float16* __restrict__ erep,   // [8192][1024] f16 (head|tail)
                      const int* __restrict__ pair,        // [32768][2]
                      const float* __restrict__ unionf,    // [32768][4096]
                      const _Float16* __restrict__ WcatT,  // [4096][1024] f16
                      const float* __restrict__ b_cat,     // [4096]
                      const _Float16* __restrict__ WrelT,  // [64][4096] f16 (zero-padded)
                      const float* __restrict__ b_rel,     // [51]
                      const float* __restrict__ freqb,     // [151*151][51]
                      const int* __restrict__ preds,       // [8192]
                      float* __restrict__ out)             // [32768][51]
{
    __shared__ __align__(16) _Float16 Alds[MT][KAP];        // ~129 KB
    __shared__ __align__(16) _Float16 Blds[2][PN][BNP];     // ~66 KB double-buffered B
    __shared__ __align__(16) _Float16 Plds[4][16][PNP];     // ~9 KB per-wave bands
    __shared__ int pair_lds[MT];

    const int tid  = threadIdx.x;
    const int wave = tid >> 5;
    const int lane = tid & 31;
    const int l15  = lane & 15;
    const int hiq  = lane >> 4;           // 0|1
    const int kb   = hiq ? 8 : 0;         // A interleave base
    const int kc   = hiq ? 8 : 0;         // B interleave base
    const int row0 = blockIdx.x * MT;

    if (tid < MT) {
        int h = pair[(size_t)(row0 + tid) * 2 + 0];
        int t = pair[(size_t)(row0 + tid) * 2 + 1];
        pair_lds[tid] = preds[h] * NOBJCLS + preds[t];
    }
    // async gather of A tile: row = concat(head_rep[h], tail_rep[t])
    for (int idx = tid; idx < MT * (KA / 8); idx += 128) {
        int r = idx >> 7;
        int k = (idx & 127) * 8;
        int src = (k < 512) ? pair[(size_t)(row0 + r) * 2 + 0]
                            : pair[(size_t)(row0 + r) * 2 + 1];
        async_b128(&erep[(size_t)src * KA + k], &Alds[r][k]);
    }
    stage_B(WcatT, Blds[0], 0, 0, tid);    // prologue: first B segment in flight too
    wait_async0();
    __syncthreads();

    const int mband = wave * 16;
    const int arow  = mband + l15;

    v8f racc[4] = {};

    for (int pc = 0; pc < POOL / PN; ++pc) {
        const int n0 = pc * PN;
        v8f pacc[4] = {};

        #pragma unroll 1
        for (int seg = 0; seg < KA / KSEG; ++seg) {
            // issue async fill of next segment (overlaps with WMMA below)
            const int nsg = pc * (KA / KSEG) + seg + 1;
            if (nsg < (POOL / PN) * (KA / KSEG)) {
                stage_B(WcatT, Blds[(seg + 1) & 1], nsg >> 2, nsg & 3, tid);
            }
            // compute on current segment
            const int buf = seg & 1;
            #pragma unroll
            for (int ks = 0; ks < KSEG; ks += 32) {
                v16h af = frag16(&Alds[arow][seg * KSEG + ks + kb],
                                 &Alds[arow][seg * KSEG + ks + 16 + kb]);
                #pragma unroll
                for (int nt = 0; nt < 4; ++nt) {
                    const _Float16* bp = &Blds[buf][nt * 16 + l15][ks + kc];
                    pacc[nt] = wmma_f16(af, frag16(bp, bp + 16), pacc[nt]);
                }
            }
            wait_async0();     // drain tail of next-segment fill
            __syncthreads();   // all waves done with this buffer / next buffer ready
        }

        // + b_post_cat, * union_features, stash f16 tile (per-wave region, no WG barrier)
        #pragma unroll
        for (int nt = 0; nt < 4; ++nt) {
            const int n = nt * 16 + l15;
            const float bias = b_cat[n0 + n];
            #pragma unroll
            for (int j = 0; j < 8; ++j) {
                const int m = j + 8 * hiq;
                const float u = unionf[(size_t)(row0 + mband + m) * POOL + n0 + n];
                Plds[wave][m][n] = (_Float16)((pacc[nt][j] + bias) * u);
            }
        }
        asm volatile("s_wait_dscnt 0x0" ::: "memory");  // intra-wave LDS visibility

        #pragma unroll
        for (int ks = 0; ks < PN; ks += 32) {
            v16h pf = frag16(&Plds[wave][l15][ks + kb], &Plds[wave][l15][ks + 16 + kb]);
            #pragma unroll
            for (int ct = 0; ct < 4; ++ct) {
                const _Float16* bp = &WrelT[(size_t)(ct * 16 + l15) * POOL + n0 + ks + kc];
                racc[ct] = wmma_f16(pf, frag16(bp, bp + 16), racc[ct]);
            }
        }
    }

    #pragma unroll
    for (int ct = 0; ct < 4; ++ct) {
        const int c = ct * 16 + l15;
        if (c < N_RELCLS) {
            #pragma unroll
            for (int j = 0; j < 8; ++j) {
                const int m   = j + 8 * hiq;
                const int row = row0 + mband + m;
                out[(size_t)row * N_RELCLS + c] =
                    racc[ct][j] + b_rel[c] + freqb[(size_t)pair_lds[mband + m] * N_RELCLS + c];
            }
        }
    }
}

// ---------- host ----------
extern "C" void kernel_launch(void* const* d_in, const int* in_sizes, int n_in,
                              void* d_out, int out_size, void* d_ws, size_t ws_size,
                              hipStream_t stream) {
    const float* edge_ctx = (const float*)d_in[0];
    const int*   pair     = (const int*)d_in[1];
    const float* unionf   = (const float*)d_in[2];
    const float* W_obj    = (const float*)d_in[3];
    const float* b_obj    = (const float*)d_in[4];
    const float* W_emb    = (const float*)d_in[5];
    const float* b_emb    = (const float*)d_in[6];
    const float* W_cat    = (const float*)d_in[7];
    const float* b_cat    = (const float*)d_in[8];
    const float* W_rel    = (const float*)d_in[9];
    const float* b_rel    = (const float*)d_in[10];
    const float* freqb    = (const float*)d_in[11];
    float* out = (float*)d_out;

    char* ws = (char*)d_ws;
    _Float16* ectx_h = (_Float16*)(ws + 0);               // 8192*512*2   = 8 MB
    _Float16* WembT  = (_Float16*)(ws + 8388608);         // 1024*512*2   = 1 MB
    _Float16* WcatT  = (_Float16*)(ws + 9437184);         // 4096*1024*2  = 8 MB
    _Float16* WrelT  = (_Float16*)(ws + 17825792);        // 64*4096*2    = 512 KB
    _Float16* erep   = (_Float16*)(ws + 18350080);        // 8192*1024*2  = 16 MB
    int*      preds  = (int*)     (ws + 35127296);        // 8192*4

    // prep
    f32_to_f16_kernel<<<(N_OBJ * HIDDEN + 255) / 256, 256, 0, stream>>>(edge_ctx, ectx_h, N_OBJ * HIDDEN);
    transpose_f16_kernel<<<(HIDDEN * 1024 + 255) / 256, 256, 0, stream>>>(W_emb, WembT, HIDDEN, 1024);
    transpose_f16_kernel<<<(1024 * POOL + 255) / 256, 256, 0, stream>>>(W_cat, WcatT, 1024, POOL);
    transpose_rel_kernel<<<(64 * POOL + 255) / 256, 256, 0, stream>>>(W_rel, WrelT);

    // obj argmax
    obj_argmax_kernel<<<N_OBJ, 256, 0, stream>>>(edge_ctx, W_obj, b_obj, preds);

    // edge_rep GEMM (WMMA)
    post_emb_kernel<<<dim3(N_OBJ / 64, 1024 / 64), 128, 0, stream>>>(ectx_h, WembT, b_emb, erep);

    // fused relation predictor (WMMA + async double-buffered B staging)
    rel_fused_kernel<<<N_REL / MT, 128, 0, stream>>>(erep, pair, unionf, WcatT, b_cat,
                                                     WrelT, b_rel, freqb, preds, out);
}
